// TransformerBlock_75110388072980
// MI455X (gfx1250) — matrix operand order repaired and checked by reading, not compiled, not run on hardware
//
#include <hip/hip_runtime.h>
#include <hip/hip_bf16.h>
#include <math.h>

// ---------------- types ----------------
typedef __bf16 bf16_t;
typedef bf16_t v16bf __attribute__((ext_vector_type(16)));
typedef bf16_t v8bf  __attribute__((ext_vector_type(8)));
typedef float  v8f   __attribute__((ext_vector_type(8)));

#define B_   2
#define S_   4096
#define D_   256
#define H_   8
#define DK_  32
#define DFF_ 1024
#define MROWS (B_*S_)   // 8192

#define MODE_QKV     0   // scatter bf16 to [B,H,S,DK]
#define MODE_ATTNOUT 1   // resid + alpha*val -> fp32
#define MODE_FFN1    2   // gelu -> bf16 row-major
#define MODE_FFN2    3   // resid + alpha*val -> fp32
#define MODE_VPACK   4   // scatter bf16 into WMMA-B packed layout

__device__ __forceinline__ bf16_t f2bf(float f) {
  union { float f; unsigned int u; } c; c.f = f;
  unsigned int u = c.u;
  unsigned int r = (u + 0x7FFFu + ((u >> 16) & 1u)) >> 16; // RNE
  union { unsigned short s; bf16_t b; } o; o.s = (unsigned short)r;
  return o.b;
}

// ---------------- fp32 -> bf16 convert (row-major) ----------------
__global__ void cvt_kernel(const float* __restrict__ src, bf16_t* __restrict__ dst, int n) {
  int i = blockIdx.x * blockDim.x + threadIdx.x;
  if (i < n) dst[i] = f2bf(src[i]);
}

// ---------------- fp32 [KxN] -> bf16 WMMA-B packed layout ----------------
// Packed layout: [K/32][N/16][lane(32)][16]; lane = (k&31)/16*16 + (n&15),
// element j = k&15.  A wave's B tile = 32 lanes x 16 contiguous bf16 (32B/lane).
__global__ void packB_kernel(const float* __restrict__ W, bf16_t* __restrict__ Wp,
                             int K, int N) {
  int i = blockIdx.x * blockDim.x + threadIdx.x;
  if (i >= K * N) return;
  int k = i / N, n = i - (i / N) * N;
  int kc = k >> 5, kk = k & 31;
  int lh = kk >> 4, j = kk & 15;
  int nt = n >> 4, lm = n & 15;
  size_t idx = ((((size_t)kc * (N >> 4) + nt) * 32) + (lh * 16 + lm)) * 16 + j;
  Wp[idx] = f2bf(W[i]);
}

// ---------------- generic bf16 WMMA GEMM (double-buffered K loop) ----------------
// A: [M x K] bf16 row-major.  Wp: B-packed bf16 (see packB).  bias: [N] fp32.
// Each wave computes a 16(M) x 64(N) strip with 4 accumulators.
// Requires K % 64 == 0 (true here: 256, 1024).
__global__ void __launch_bounds__(256)
gemm_kernel(const bf16_t* __restrict__ A, const bf16_t* __restrict__ Wp,
            const float* __restrict__ bias, int M, int N, int K, int mode,
            bf16_t* __restrict__ outb, float* __restrict__ outf,
            const float* __restrict__ resid, const float* __restrict__ alphap)
{
  const int lane  = threadIdx.x & 31;
  const int wid   = threadIdx.x >> 5;
  const int mtile = blockIdx.x * 8 + wid;
  const int mbase = mtile * 16;
  const int nbase = blockIdx.y * 64;
  if (mbase >= M) return;

  const int lmod  = lane & 15;
  const int lhalf = lane >> 4;
  const int ntile0 = nbase >> 4;
  const size_t bstride = (size_t)(N >> 4) * 512;  // elements per K-chunk of packed B

  v8f acc[4];
  #pragma unroll
  for (int t = 0; t < 4; ++t)
    #pragma unroll
    for (int r = 0; r < 8; ++r) acc[t][r] = 0.f;

  const int arow = mbase + lmod;
  const bf16_t* arowp = A + (size_t)arow * K;
  const bf16_t* blane = Wp + ((size_t)ntile0 * 32 + lane) * 16;

  auto loadChunk = [&](int c, v16bf& a, v16bf* b) {
    const v8bf* ap = (const v8bf*)(arowp + (c << 5) + lhalf * 16);
    v8bf lo = ap[0];   // K = 32c + h*16 .. +7
    v8bf hi = ap[1];   // K = 32c + h*16 + 8 .. +15
    #pragma unroll
    for (int i = 0; i < 8; ++i) { a[i] = lo[i]; a[8 + i] = hi[i]; }
    const bf16_t* bc = blane + (size_t)c * bstride;
    #pragma unroll
    for (int t = 0; t < 4; ++t) b[t] = *(const v16bf*)(bc + (size_t)t * 512);
  };
  auto domm = [&](const v16bf& a, const v16bf* b) {
    #pragma unroll
    for (int t = 0; t < 4; ++t)
      acc[t] = __builtin_amdgcn_wmma_f32_16x16x32_bf16(
                 false, a, false, b[t], (short)0, acc[t], false, false);
  };

  v16bf aA, aB;
  v16bf bA[4], bB[4];
  loadChunk(0, aA, bA);
  const int nch = K >> 5;     // even
  int c = 0;
  for (; c + 2 < nch; c += 2) {
    loadChunk(c + 1, aB, bB);
    __builtin_prefetch((const void*)(arowp + ((c + 3) << 5)), 0, 3);
    __builtin_prefetch((const void*)(blane + (size_t)(c + 3) * bstride), 0, 3);
    domm(aA, bA);
    loadChunk(c + 2, aA, bA);
    domm(aB, bB);
  }
  loadChunk(c + 1, aB, bB);
  domm(aA, bA);
  domm(aB, bB);

  const float alpha = alphap ? alphap[0] : 0.f;
  #pragma unroll
  for (int t = 0; t < 4; ++t) {
    const int col = nbase + t * 16 + lmod;
    const float bv = bias[col];
    #pragma unroll
    for (int r = 0; r < 8; ++r) {
      const int row = mbase + r + lhalf * 8;
      float val = acc[t][r] + bv;
      if (mode == MODE_QKV) {
        int bi = row / S_;
        int s  = row - bi * S_;
        int h  = col >> 5;
        int dk = col & 31;
        outb[(((size_t)bi * H_ + h) * S_ + s) * DK_ + dk] = f2bf(val);
      } else if (mode == MODE_VPACK) {
        // pack V directly into WMMA-B layout per (bh, key-chunk, dk-tile)
        int bi = row / S_;
        int s  = row - bi * S_;          // key
        int h  = col >> 5;
        int dk = col & 31;
        int bh = bi * H_ + h;
        int kc = s >> 5, kk = s & 31;
        int lh2 = kk >> 4, j2 = kk & 15;
        int t2 = dk >> 4, lm2 = dk & 15;
        size_t idx = ((((size_t)bh * (S_ / 32) + kc) * 2 + t2) * 32 + (lh2 * 16 + lm2)) * 16 + j2;
        outb[idx] = f2bf(val);
      } else if (mode == MODE_FFN1) {
        float g = 0.5f * val * (1.0f + erff(val * 0.70710678118654752f));
        outb[(size_t)row * N + col] = f2bf(g);
      } else { // MODE_ATTNOUT / MODE_FFN2: residual + alpha * val
        outf[(size_t)row * N + col] = resid[(size_t)row * N + col] + alpha * val;
      }
    }
  }
}

// ---------------- flash attention ----------------
// One wave per (b, h, 16-query tile). Q/K bf16 [B,H,S,DK]; Vp packed; ctx bf16 [B,S,D].
__global__ void __launch_bounds__(256)
flash_kernel(const bf16_t* __restrict__ Q, const bf16_t* __restrict__ Kk,
             const bf16_t* __restrict__ Vp, const int* __restrict__ mask,
             bf16_t* __restrict__ ctx)
{
  __shared__ __align__(16) bf16_t smem[8 * 16 * 32];
  const int lane   = threadIdx.x & 31;
  const int wid    = threadIdx.x >> 5;
  const int gw     = blockIdx.x * 8 + wid;
  const int qtiles = S_ / 16;
  const int bh     = gw / qtiles;
  const int qt     = gw % qtiles;
  const int bi     = bh / H_;
  const int h      = bh % H_;
  const int lmod   = lane & 15;
  const int lhalf  = lane >> 4;
  bf16_t* lds = smem + wid * (16 * 32);

  // Q tile in A-layout (16 x DK=32)
  v16bf qa;
  {
    const bf16_t* qb = Q + ((size_t)bh * S_ + qt * 16) * DK_;
    const v8bf* qp = (const v8bf*)(qb + (size_t)lmod * DK_ + lhalf * 16);
    v8bf lo = qp[0], hi = qp[1];
    #pragma unroll
    for (int i = 0; i < 8; ++i) { qa[i] = lo[i]; qa[8 + i] = hi[i]; }
  }

  float mrow[8], lrow[8];
  v8f acc0, acc1;
  #pragma unroll
  for (int r = 0; r < 8; ++r) { mrow[r] = -3.0e38f; lrow[r] = 0.f; acc0[r] = 0.f; acc1[r] = 0.f; }

  const float scale = 0.17677669529663687f; // 1/sqrt(32)
  const bf16_t* kbase = Kk + (size_t)bh * S_ * DK_;
  const bf16_t* vbase = Vp + (size_t)bh * ((S_ / 32) * 1024); // per-bh packed V
  const int*    mb    = mask + (size_t)bi * S_;

  for (int kb = 0; kb < S_; kb += 32) {
    // ================= front-load ALL memory for this key chunk =================
    // K^T B-tiles: K-dim = dk, col = key (two 16B loads per lane)
    const v8bf* p0 = (const v8bf*)(kbase + (size_t)(kb + lmod) * DK_ + lhalf * 16);
    const v8bf* p1 = (const v8bf*)(kbase + (size_t)(kb + 16 + lmod) * DK_ + lhalf * 16);
    v8bf a0 = p0[0], a1 = p0[1], c0 = p1[0], c1 = p1[1];
    // packed V B-tiles (used only after softmax -> long latency cover)
    const bf16_t* vchunk = vbase + (size_t)(kb >> 5) * 1024 + (size_t)lane * 16;
    v16bf bv0 = *(const v16bf*)(vchunk);
    v16bf bv1 = *(const v16bf*)(vchunk + 512);
    // mask words
    const bool z0 = (mb[kb + lmod] == 0);
    const bool z1 = (mb[kb + 16 + lmod] == 0);
    // prefetch next chunk
    __builtin_prefetch((const void*)(kbase + (size_t)(kb + 32 + lmod) * DK_), 0, 3);
    __builtin_prefetch((const void*)(vchunk + 1024), 0, 3);

    v16bf bk0, bk1;
    #pragma unroll
    for (int j = 0; j < 8; ++j) {
      bk0[j] = a0[j]; bk0[8 + j] = a1[j];
      bk1[j] = c0[j]; bk1[8 + j] = c1[j];
    }

    // ================= scores =================
    v8f s0, s1;
    #pragma unroll
    for (int r = 0; r < 8; ++r) { s0[r] = 0.f; s1[r] = 0.f; }
    s0 = __builtin_amdgcn_wmma_f32_16x16x32_bf16(false, qa, false, bk0, (short)0, s0, false, false);
    s1 = __builtin_amdgcn_wmma_f32_16x16x32_bf16(false, qa, false, bk1, (short)0, s1, false, false);

    #pragma unroll
    for (int r = 0; r < 8; ++r) {
      s0[r] = z0 ? -1e9f : s0[r] * scale;
      s1[r] = z1 ? -1e9f : s1[r] * scale;
    }

    // ================= online softmax (rows per vgpr slot x lane-half) =========
    #pragma unroll
    for (int r = 0; r < 8; ++r) {
      float rm = fmaxf(s0[r], s1[r]);
      #pragma unroll
      for (int o = 1; o < 16; o <<= 1) rm = fmaxf(rm, __shfl_xor(rm, o, 16));
      float mn = fmaxf(mrow[r], rm);
      float f  = __expf(mrow[r] - mn);
      float p0e = __expf(s0[r] - mn);
      float p1e = __expf(s1[r] - mn);
      s0[r] = p0e; s1[r] = p1e;
      float rs = p0e + p1e;
      #pragma unroll
      for (int o = 1; o < 16; o <<= 1) rs += __shfl_xor(rs, o, 16);
      lrow[r] = lrow[r] * f + rs;
      mrow[r] = mn;
      acc0[r] *= f; acc1[r] *= f;
    }

    // ================= P: C-layout -> LDS -> A-layout (wave-private) ===========
    #pragma unroll
    for (int r = 0; r < 8; ++r) {
      const int row = r + lhalf * 8;
      lds[row * 32 + lmod]      = f2bf(s0[r]);
      lds[row * 32 + 16 + lmod] = f2bf(s1[r]);
    }
    asm volatile("s_wait_dscnt 0x0" ::: "memory");
    v16bf pa;
    {
      const v8bf* pp = (const v8bf*)(lds + lmod * 32 + lhalf * 16);
      v8bf lo = pp[0], hi = pp[1];
      #pragma unroll
      for (int i = 0; i < 8; ++i) { pa[i] = lo[i]; pa[8 + i] = hi[i]; }
    }

    // ================= context accumulate ======================================
    acc0 = __builtin_amdgcn_wmma_f32_16x16x32_bf16(false, pa, false, bv0, (short)0, acc0, false, false);
    acc1 = __builtin_amdgcn_wmma_f32_16x16x32_bf16(false, pa, false, bv1, (short)0, acc1, false, false);
  }

  // ---- normalize + store ctx bf16 [B,S,D] ----
  #pragma unroll
  for (int r = 0; r < 8; ++r) {
    const int srow = qt * 16 + r + lhalf * 8;
    const float inv = 1.0f / lrow[r];
    const size_t o = ((size_t)bi * S_ + srow) * D_ + h * DK_;
    ctx[o + lmod]      = f2bf(acc0[r] * inv);
    ctx[o + 16 + lmod] = f2bf(acc1[r] * inv);
  }
}

// ---------------- layernorm (one row per block) ----------------
__global__ void __launch_bounds__(256)
ln_kernel(const float* __restrict__ in, const float* __restrict__ g,
          const float* __restrict__ be, float* __restrict__ outf,
          bf16_t* __restrict__ outb)
{
  __shared__ float red[256];
  const int row = blockIdx.x;
  const int tid = threadIdx.x;
  float v = in[(size_t)row * D_ + tid];
  red[tid] = v; __syncthreads();
  for (int s = 128; s > 0; s >>= 1) { if (tid < s) red[tid] += red[tid + s]; __syncthreads(); }
  const float mu = red[0] / (float)D_;
  __syncthreads();
  const float d = v - mu;
  red[tid] = d * d; __syncthreads();
  for (int s = 128; s > 0; s >>= 1) { if (tid < s) red[tid] += red[tid + s]; __syncthreads(); }
  const float var = red[0] / (float)D_;
  const float y = d * rsqrtf(var + 1e-5f) * g[tid] + be[tid];
  outf[(size_t)row * D_ + tid] = y;
  if (outb) outb[(size_t)row * D_ + tid] = f2bf(y);
}

// ---------------- host orchestration ----------------
extern "C" void kernel_launch(void* const* d_in, const int* in_sizes, int n_in,
                              void* d_out, int out_size, void* d_ws, size_t ws_size,
                              hipStream_t stream) {
  (void)in_sizes; (void)n_in; (void)out_size; (void)ws_size;
  const float* x   = (const float*)d_in[0];
  const int*   msk = (const int*)  d_in[1];
  const float* Wq  = (const float*)d_in[2];  const float* bq = (const float*)d_in[3];
  const float* Wk  = (const float*)d_in[4];  const float* bk = (const float*)d_in[5];
  const float* Wv  = (const float*)d_in[6];  const float* bv = (const float*)d_in[7];
  const float* Wo  = (const float*)d_in[8];  const float* bo = (const float*)d_in[9];
  const float* W1  = (const float*)d_in[10]; const float* b1 = (const float*)d_in[11];
  const float* W2  = (const float*)d_in[12]; const float* b2 = (const float*)d_in[13];
  const float* g1  = (const float*)d_in[14]; const float* be1 = (const float*)d_in[15];
  const float* g2  = (const float*)d_in[16]; const float* be2 = (const float*)d_in[17];
  const float* a1  = (const float*)d_in[18]; const float* a2  = (const float*)d_in[19];

  char* ws = (char*)d_ws;
  size_t off = 0;
  auto alloc = [&](size_t bytes) -> void* {
    void* p = ws + off; off += (bytes + 255) & ~(size_t)255; return p;
  };
  bf16_t* Wqp = (bf16_t*)alloc((size_t)D_ * D_ * 2);
  bf16_t* Wkp = (bf16_t*)alloc((size_t)D_ * D_ * 2);
  bf16_t* Wvp = (bf16_t*)alloc((size_t)D_ * D_ * 2);
  bf16_t* Wop = (bf16_t*)alloc((size_t)D_ * D_ * 2);
  bf16_t* W1p = (bf16_t*)alloc((size_t)D_ * DFF_ * 2);
  bf16_t* W2p = (bf16_t*)alloc((size_t)DFF_ * D_ * 2);
  bf16_t* xb  = (bf16_t*)alloc((size_t)MROWS * D_ * 2);
  bf16_t* Qb  = (bf16_t*)alloc((size_t)B_ * H_ * S_ * DK_ * 2);
  bf16_t* Kb  = (bf16_t*)alloc((size_t)B_ * H_ * S_ * DK_ * 2);
  bf16_t* Vp  = (bf16_t*)alloc((size_t)B_ * H_ * S_ * DK_ * 2); // packed
  bf16_t* ctxb= (bf16_t*)alloc((size_t)MROWS * D_ * 2);
  float*  y1  = (float*) alloc((size_t)MROWS * D_ * 4);
  float*  x1  = (float*) alloc((size_t)MROWS * D_ * 4);
  bf16_t* x1b = (bf16_t*)alloc((size_t)MROWS * D_ * 2);
  bf16_t* hb  = (bf16_t*)alloc((size_t)MROWS * DFF_ * 2);
  float*  y2  = (float*) alloc((size_t)MROWS * D_ * 4);

  // activations: row-major bf16 convert
  cvt_kernel<<<(MROWS * D_ + 255) / 256, 256, 0, stream>>>(x, xb, MROWS * D_);
  // weights: pack into WMMA-B layout
  auto pack = [&](const float* s, bf16_t* d, int K, int N) {
    packB_kernel<<<(K * N + 255) / 256, 256, 0, stream>>>(s, d, K, N);
  };
  pack(Wq, Wqp, D_, D_);  pack(Wk, Wkp, D_, D_);
  pack(Wv, Wvp, D_, D_);  pack(Wo, Wop, D_, D_);
  pack(W1, W1p, D_, DFF_); pack(W2, W2p, DFF_, D_);

  dim3 gD(MROWS / 128, D_ / 64);     // 64 x 4
  dim3 gF(MROWS / 128, DFF_ / 64);   // 64 x 16

  // QKV projections
  gemm_kernel<<<gD, 256, 0, stream>>>(xb, Wqp, bq, MROWS, D_, D_, MODE_QKV,   Qb, nullptr, nullptr, nullptr);
  gemm_kernel<<<gD, 256, 0, stream>>>(xb, Wkp, bk, MROWS, D_, D_, MODE_QKV,   Kb, nullptr, nullptr, nullptr);
  gemm_kernel<<<gD, 256, 0, stream>>>(xb, Wvp, bv, MROWS, D_, D_, MODE_VPACK, Vp, nullptr, nullptr, nullptr);

  // flash attention -> ctx bf16 [B,S,D]
  flash_kernel<<<(B_ * H_ * (S_ / 16)) / 8, 256, 0, stream>>>(Qb, Kb, Vp, msk, ctxb);

  // output projection + residual (y1 = x + a1*attn_out)
  gemm_kernel<<<gD, 256, 0, stream>>>(ctxb, Wop, bo, MROWS, D_, D_, MODE_ATTNOUT, nullptr, y1, x, a1);
  ln_kernel<<<MROWS, 256, 0, stream>>>(y1, g1, be1, x1, x1b);

  // FFN
  gemm_kernel<<<gF, 256, 0, stream>>>(x1b, W1p, b1, MROWS, DFF_, D_, MODE_FFN1, hb, nullptr, nullptr, nullptr);
  gemm_kernel<<<gD, 256, 0, stream>>>(hb, W2p, b2, MROWS, D_, DFF_, MODE_FFN2, nullptr, y2, x1, a2);
  ln_kernel<<<MROWS, 256, 0, stream>>>(y2, g2, be2, (float*)d_out, nullptr);
}
